// snn_layer_75076028334718
// MI455X (gfx1250) — matrix operand order repaired and checked by reading, not compiled, not run on hardware
//
#include <hip/hip_runtime.h>

// CDNA5 / gfx1250 fused SNN layer:
//   h = einsum('bft,fu->btu', x, W)  via bf16 WMMA (16x16x32),
//   then in-register leaky integrate-and-fire scan over t -> spikes.
// x: [B=256, F=784, T=128] f32, W: [F=784, U=2048] f32, out: [B,T,U] f32.
//
// Pipeline per K-chunk (32 wide, 784 padded to 800):
//   fetch next chunk (global b128 -> regs) | WMMA current chunk from LDS |
//   convert+store fetched regs -> other LDS buffer | one barrier.

typedef __attribute__((ext_vector_type(16))) __bf16 v16bf;
typedef __attribute__((ext_vector_type(2)))  __bf16 v2bf;
typedef __attribute__((ext_vector_type(8)))  float  v8f;

#define F_DIM 784
#define T_DIM 128
#define U_DIM 2048
#define KT    32          // K per WMMA step (bf16)
#define NK    25          // ceil(784/32); last chunk has 16 valid rows
#define PAD   40          // LDS row stride in ushorts (80B: 16B-aligned,
                          // 20-dword stride => conflict-free b128 operand reads)

__device__ __forceinline__ unsigned short f2bf(float f) {
  union { float f; unsigned int u; } c; c.f = f;
  unsigned int u = c.u;
  u += 0x7FFFu + ((u >> 16) & 1u);   // RNE f32 -> bf16
  return (unsigned short)(u >> 16);
}

// Pack two f32 -> one dword of two bf16 (lo, hi). Native V_CVT_PK_BF16_F32
// when the toolchain declares it; bit-twiddled RNE fallback otherwise.
__device__ __forceinline__ unsigned int pack_bf16(float lo, float hi) {
#if __has_builtin(__builtin_amdgcn_cvt_pk_bf16_f32)
  union { v2bf v; unsigned int u; } c;
  c.v = __builtin_amdgcn_cvt_pk_bf16_f32(lo, hi);
  return c.u;
#else
  return (unsigned int)f2bf(lo) | ((unsigned int)f2bf(hi) << 16);
#endif
}

struct Stage {             // per-thread register staging of one K-chunk
  float4 a[2][2];          // [it][row of f-pair], 4 t's each
  float4 b[2][2];          // [it][row of f-pair], 4 u's each
};

// Each thread covers t/u = (tid&31)*4 .. +3 and f-pairs fp=(tid>>5)*2 (+16).
__device__ __forceinline__ void fetchChunk(Stage& s, const float* __restrict__ xb,
                                           const float* __restrict__ wb,
                                           int k0, int tq, int fp, bool tail) {
#pragma unroll
  for (int it = 0; it < 2; ++it) {
    const int fl = fp + it * 16;     // even, 0..30; tail chunk: only fl<16 valid
    if (tail && it == 1) {
      s.a[it][0] = float4{0.f, 0.f, 0.f, 0.f};
      s.a[it][1] = float4{0.f, 0.f, 0.f, 0.f};
      s.b[it][0] = float4{0.f, 0.f, 0.f, 0.f};
      s.b[it][1] = float4{0.f, 0.f, 0.f, 0.f};
    } else {
      s.a[it][0] = *(const float4*)(xb + (size_t)(k0 + fl) * T_DIM + tq);
      s.a[it][1] = *(const float4*)(xb + (size_t)(k0 + fl + 1) * T_DIM + tq);
      s.b[it][0] = *(const float4*)(wb + (size_t)(k0 + fl) * U_DIM + tq);
      s.b[it][1] = *(const float4*)(wb + (size_t)(k0 + fl + 1) * U_DIM + tq);
    }
  }
}

__device__ __forceinline__ void storeChunk(const Stage& s, unsigned short* la,
                                           unsigned short* lb, int tq, int fp) {
#pragma unroll
  for (int it = 0; it < 2; ++it) {
    const int fl = fp + it * 16;
    const float* a0 = (const float*)&s.a[it][0];
    const float* a1 = (const float*)&s.a[it][1];
    const float* b0 = (const float*)&s.b[it][0];
    const float* b1 = (const float*)&s.b[it][1];
#pragma unroll
    for (int j = 0; j < 4; ++j) {    // (k, k+1) bf16 pair -> one dword
      *(unsigned int*)(la + (tq + j) * PAD + fl) = pack_bf16(a0[j], a1[j]);
      *(unsigned int*)(lb + (tq + j) * PAD + fl) = pack_bf16(b0[j], b1[j]);
    }
  }
}

union U8 { unsigned int u[8]; v16bf v; };

__device__ __forceinline__ void computeChunk(v8f acc[8],
                                             const unsigned short* la,
                                             const unsigned short* lb,
                                             int n, int half, int ucol) {
  // B operand: lane=N, VGPR 0..3: K = half*8 + 2r, VGPR 4..7: K = 16 + half*8 + 2r.
  U8 bb;
  const unsigned short* bp = lb + ucol * PAD + half * 8;
#pragma unroll
  for (int r = 0; r < 8; ++r)
    bb.u[r] = *(const unsigned int*)(bp + (r >> 2) * 16 + 2 * (r & 3));

  // Preload all 8 A operands (lane holds row M=n; VGPR 0..3: K=half*8+2r,
  // VGPR 4..7: K=16+half*8+2r) so ds_loads issue ahead of the WMMA burst.
  U8 aa[8];
#pragma unroll
  for (int i = 0; i < 8; ++i) {
    const unsigned short* ap = la + (i * 16 + n) * PAD + half * 8;
#pragma unroll
    for (int r = 0; r < 4; ++r) aa[i].u[r]     = *(const unsigned int*)(ap + 2 * r);
#pragma unroll
    for (int r = 0; r < 4; ++r) aa[i].u[4 + r] = *(const unsigned int*)(ap + 16 + 2 * r);
  }
#pragma unroll
  for (int i = 0; i < 8; ++i)
    acc[i] = __builtin_amdgcn_wmma_f32_16x16x32_bf16(
        false, aa[i].v, false, bb.v, (short)0, acc[i], false, false);
}

__global__ __launch_bounds__(256)
void snn_fused_wmma(const float* __restrict__ x,
                    const float* __restrict__ W,
                    float* __restrict__ out) {
  __shared__ unsigned short lds_a[2][T_DIM * PAD];   // 2 x 10 KB
  __shared__ unsigned short lds_b[2][T_DIM * PAD];   // 2 x 10 KB

  const int tid  = threadIdx.x;
  const int lane = tid & 31;
  const int wv   = tid >> 5;
  const int n    = lane & 15;
  const int half = lane >> 4;
  const int ucol = wv * 16 + n;
  const int tq   = (tid & 31) * 4;   // staging: 4 consecutive t (and u)
  const int fp   = (tid >> 5) * 2;   // staging: f-pair base

  const int b  = blockIdx.y;
  const int u0 = blockIdx.x * 128;

  const float* xb = x + (size_t)b * F_DIM * T_DIM;
  const float* wb = W + u0;

  v8f acc[8];
#pragma unroll
  for (int i = 0; i < 8; ++i)
#pragma unroll
    for (int j = 0; j < 8; ++j) acc[i][j] = 0.0f;

  // Prologue: stage chunk 0.
  Stage st;
  fetchChunk(st, xb, wb, 0, tq, fp, false);
  storeChunk(st, lds_a[0], lds_b[0], tq, fp);
  __syncthreads();

  for (int kc = 0; kc < NK; ++kc) {
    const int cur = kc & 1;
    const bool last = (kc == NK - 1);
    if (!last)
      fetchChunk(st, xb, wb, (kc + 1) * KT, tq, fp, (kc + 1) == NK - 1);
    computeChunk(acc, lds_a[cur], lds_b[cur], n, half, ucol);
    if (!last) {
      storeChunk(st, lds_a[cur ^ 1], lds_b[cur ^ 1], tq, fp);
      __syncthreads();
    }
  }

  // In-register LIF scan over t. C layout: VGPR r <-> M = r + 8*half; state
  // ping-pongs between lane pair (L, L^16) every 8 steps via wave32 shuffles.
  float syn = 0.0f, mem = 0.0f;
#pragma unroll
  for (int i = 0; i < 8; ++i) {
    if (half == 0) {
#pragma unroll
      for (int r = 0; r < 8; ++r) {
        syn = 0.9f  * syn + acc[i][r];
        mem = 0.85f * mem + syn;
        const float spk = (mem > 1.0f) ? 1.0f : 0.0f;
        mem -= spk;                  // soft reset by threshold (=1.0)
        acc[i][r] = spk;
      }
    }
    syn = __shfl_xor(syn, 16, 32);
    mem = __shfl_xor(mem, 16, 32);
    if (half == 1) {
#pragma unroll
      for (int r = 0; r < 8; ++r) {
        syn = 0.9f  * syn + acc[i][r];
        mem = 0.85f * mem + syn;
        const float spk = (mem > 1.0f) ? 1.0f : 0.0f;
        mem -= spk;
        acc[i][r] = spk;
      }
    }
    syn = __shfl_xor(syn, 16, 32);
    mem = __shfl_xor(mem, 16, 32);
  }

  // Store spikes directly: out[b, t, u0+ucol].
  float* ob = out + (size_t)b * T_DIM * U_DIM + u0 + ucol;
#pragma unroll
  for (int i = 0; i < 8; ++i)
#pragma unroll
    for (int r = 0; r < 8; ++r) {
      const int t = i * 16 + half * 8 + r;
      ob[(size_t)t * U_DIM] = acc[i][r];
    }
}

extern "C" void kernel_launch(void* const* d_in, const int* in_sizes, int n_in,
                              void* d_out, int out_size, void* d_ws, size_t ws_size,
                              hipStream_t stream) {
  (void)in_sizes; (void)n_in; (void)out_size; (void)d_ws; (void)ws_size;
  const float* x = (const float*)d_in[0];   // [256, 784, 128]
  const float* W = (const float*)d_in[1];   // [784, 2048]
  float* out = (float*)d_out;               // [256, 128, 2048]

  dim3 grid(U_DIM / 128, 256);              // (u tiles, batch)
  dim3 block(256);                          // 8 wave32
  hipLaunchKernelGGL(snn_fused_wmma, grid, block, 0, stream, x, W, out);
}
